// SkipRefinement_19155554140426
// MI455X (gfx1250) — matrix operand order repaired and checked by reading, not compiled, not run on hardware
//
#include <hip/hip_runtime.h>
#include <hip/hip_fp16.h>

// ---------------- CDNA5 / problem constants ----------------
typedef __attribute__((ext_vector_type(16))) _Float16 v16h;
typedef __attribute__((ext_vector_type(4)))  _Float16 v4h;
typedef __attribute__((ext_vector_type(8)))  float    v8f;

#define C_DIM   256
#define NH      8
#define DH      32          // head dim == WMMA K step for f16 (16x16x32)
#define WINSZ   6
#define NTOK    216         // 6*6*6
#define NPAD    224         // 14 tiles of 16
#define NB      2
#define DIM3    48
#define DHW     (48*48*48)
#define NWIN_B  512         // (48/6)^3
#define BW      1024        // NB * NWIN_B
#define LN_EPS  1e-5f
// scale folded into q LayerNorm: 1/sqrt(32) * log2(e)  ->  softmax uses exp2
#define Q_FOLD_SCALE (0.17677669529663687f * 1.4426950408889634f)
#define OUT_LDS_STRIDE 217  // 216 padded +1 dword: coprime with 64 banks
#define TPB 224             // 7 waves: 14 M-tiles / 7 waves = exactly 2 each

__device__ __forceinline__ int a_kidx(int e, int half) {
    // 16-bit A-matrix 16x32 layout: contiguous k = 16*half + e (matches B)
    return e + 16 * half;
}

// CDNA5 async DMA: memory -> LDS, 16B chunk (ASYNCcnt tracked)
__device__ __forceinline__ void async_copy_b128(unsigned ldsaddr, const void* gaddr) {
    asm volatile("global_load_async_to_lds_b128 %0, %1, off"
                 :: "v"(ldsaddr), "v"(gaddr) : "memory");
}
// CDNA5 async DMA: LDS -> memory, 4B chunk (ASYNCcnt tracked)
__device__ __forceinline__ void async_store_b32(void* gaddr, unsigned ldsaddr) {
    asm volatile("global_store_async_from_lds_b32 %0, %1, off"
                 :: "v"(gaddr), "v"(ldsaddr) : "memory");
}
__device__ __forceinline__ void wait_asynccnt0() {
    asm volatile("s_wait_asynccnt 0" ::: "memory");
}

// =====================================================================
// Kernel 1: fused LayerNorm + 3D window partition, f32 -> f16.
// q additionally pre-scaled by 1/sqrt(dh)*log2(e) so attention softmax
// needs no multiplies. grid (864, 3); reads coalesced along W.
// =====================================================================
__global__ void ln_partition_kernel(const float* __restrict__ qm,
                                    const float* __restrict__ km,
                                    const float* __restrict__ vm,
                                    const float* __restrict__ wq,
                                    const float* __restrict__ bq,
                                    const float* __restrict__ wkv,
                                    const float* __restrict__ bkv,
                                    _Float16* __restrict__ qn,
                                    _Float16* __restrict__ kn,
                                    _Float16* __restrict__ vn) {
    int map = blockIdx.y;
    const float* src_map = (map == 0) ? qm : (map == 1 ? km : vm);
    const float* lw = (map == 0) ? wq : wkv;
    const float* lb = (map == 0) ? bq : bkv;
    _Float16* dst = (map == 0) ? qn : (map == 1 ? kn : vn);
    float fold = (map == 0) ? Q_FOLD_SCALE : 1.0f;

    int sg = blockIdx.x * blockDim.x + threadIdx.x;
    if (sg >= NB * DHW) return;
    int b = sg / DHW;
    int s = sg - b * DHW;
    int d = s / (DIM3 * DIM3);
    int rem = s - d * DIM3 * DIM3;
    int h = rem / DIM3;
    int w = rem - h * DIM3;

    int win = b * NWIN_B + ((d / WINSZ) * 8 + (h / WINSZ)) * 8 + (w / WINSZ);
    int tok = (d % WINSZ) * 36 + (h % WINSZ) * 6 + (w % WINSZ);

    const float* src = src_map + (size_t)b * C_DIM * DHW + s;

    float sum = 0.f, ssq = 0.f;
    for (int c = 0; c < C_DIM; ++c) {
        float v = src[(size_t)c * DHW];   // coalesced across wave
        sum += v; ssq += v * v;
    }
    float mean = sum * (1.f / C_DIM);
    float var  = ssq * (1.f / C_DIM) - mean * mean;
    float rstd = rsqrtf(var + LN_EPS) * fold;
    float bofs = fold;   // bias also scaled

    _Float16* drow = dst + ((size_t)win * NTOK + tok) * C_DIM;
    for (int c = 0; c < C_DIM; c += 4) {   // 8B packed stores
        v4h pk;
        #pragma unroll
        for (int j = 0; j < 4; ++j) {
            float v = src[(size_t)(c + j) * DHW];   // L2 hit
            pk[j] = (_Float16)((v - mean) * rstd * lw[c + j] + lb[c + j] * bofs);
        }
        *(v4h*)(drow + c) = pk;
    }
}

// =====================================================================
// Kernel 2: projection weight f32 -> f16 (128 KB, stays L2 resident)
// =====================================================================
__global__ void cvt_w_kernel(const float* __restrict__ pw,
                             _Float16* __restrict__ pw16, int n) {
    int i = blockIdx.x * blockDim.x + threadIdx.x;
    if (i < n) pw16[i] = (_Float16)pw[i];
}

// =====================================================================
// Kernel 3: fused window attention. One block = (window, head), 7 waves.
// Staging via GLOBAL_LOAD_ASYNC_TO_LDS_B128; all matmuls via
// v_wmma_f32_16x16x32_f16 (K=32 == head dim: 1 WMMA per score tile).
// Softmax: exp2 (scale pre-folded), 1/sum deferred to output.
// =====================================================================
__global__ void __launch_bounds__(TPB)
attn_kernel(const _Float16* __restrict__ qn,
            const _Float16* __restrict__ kn,
            const _Float16* __restrict__ vn,
            _Float16* __restrict__ ows) {
    extern __shared__ char smem[];
    _Float16* k_lds = (_Float16*)smem;            // [NPAD][DH]
    _Float16* v_lds = k_lds + NPAD * DH;          // [NPAD][DH]
    _Float16* q_lds = v_lds + NPAD * DH;          // [NPAD][DH]
    _Float16* p_lds = q_lds + NPAD * DH;          // [7 waves][16][NPAD]

    int win  = blockIdx.x;
    int head = blockIdx.y;
    int tid  = threadIdx.x;
    const size_t base = (size_t)win * NTOK * C_DIM + (size_t)head * DH;

    // ---- async DMA the 216 real rows (64B/token = 4 x b128 chunks) ----
    {
        unsigned lq = (unsigned)(size_t)q_lds;  // wave-relative LDS byte addr
        unsigned lk = (unsigned)(size_t)k_lds;
        unsigned lv = (unsigned)(size_t)v_lds;
        for (int i = tid; i < NTOK * 4; i += TPB) {
            int t = i >> 2, c16 = i & 3;
            unsigned loff = (unsigned)(t * DH * 2 + c16 * 16);
            size_t   goff = base + (size_t)t * C_DIM + c16 * 8;
            async_copy_b128(lq + loff, qn + goff);
            async_copy_b128(lk + loff, kn + goff);
            async_copy_b128(lv + loff, vn + goff);
        }
        // zero-fill padded rows [216,224) while DMA is in flight
        for (int i = tid; i < (NPAD - NTOK) * DH; i += TPB) {
            q_lds[NTOK * DH + i] = (_Float16)0.f;
            k_lds[NTOK * DH + i] = (_Float16)0.f;
            v_lds[NTOK * DH + i] = (_Float16)0.f;
        }
        wait_asynccnt0();
    }
    __syncthreads();

    int wave = tid >> 5, lane = tid & 31;
    int half = lane >> 4, lrow = lane & 15;
    _Float16* pw = p_lds + wave * 16 * NPAD;

    for (int mt = wave; mt < 14; mt += 7) {
        // ---- A fragment: Q rows (K = 32 = full head dim) ----
        v16h aq;
        #pragma unroll
        for (int e = 0; e < 16; ++e)
            aq[e] = q_lds[(mt * 16 + lrow) * DH + a_kidx(e, half)];

        // ---- S = Q K^T : one WMMA per 16x16 tile ----
        v8f sc[14];
        v8f zero = {};
        #pragma unroll
        for (int nt = 0; nt < 14; ++nt) {
            v16h bk;  // B[k][n] = K[n][k]
            #pragma unroll
            for (int e = 0; e < 16; ++e)
                bk[e] = k_lds[(nt * 16 + lrow) * DH + (e + 16 * half)];
            sc[nt] = __builtin_amdgcn_wmma_f32_16x16x32_f16(
                false, aq, false, bk, (short)0, zero, false, false);
        }

        // ---- row softmax (scores already in log2 domain; mask >= 216) ----
        float invr[8];
        #pragma unroll
        for (int r = 0; r < 8; ++r) {
            float mx = -1e30f;
            #pragma unroll
            for (int nt = 0; nt < 14; ++nt) {
                int col = nt * 16 + lrow;
                float v = (col < NTOK) ? sc[nt][r] : -1e30f;  // only nt=13 costs
                sc[nt][r] = v;
                mx = fmaxf(mx, v);
            }
            #pragma unroll
            for (int off = 8; off >= 1; off >>= 1)
                mx = fmaxf(mx, __shfl_xor(mx, off, 32));   // 16-lane groups
            float sum = 0.f;
            #pragma unroll
            for (int nt = 0; nt < 14; ++nt) {
                float e = exp2f(sc[nt][r] - mx);           // v_exp_f32
                sc[nt][r] = e; sum += e;
            }
            #pragma unroll
            for (int off = 8; off >= 1; off >>= 1)
                sum += __shfl_xor(sum, off, 32);
            invr[r] = 1.f / sum;                           // applied after PV
            int row = r + 8 * half;                        // C-layout row id
            #pragma unroll
            for (int nt = 0; nt < 14; ++nt)                // unnormalized P
                pw[row * NPAD + nt * 16 + lrow] = (_Float16)sc[nt][r];
        }

        // ---- O = P V : 7 K-steps, 2 output n-tiles (dh = 32) ----
        v8f oc0 = {}, oc1 = {};
        #pragma unroll
        for (int kk = 0; kk < 7; ++kk) {
            v16h ap;
            #pragma unroll
            for (int e = 0; e < 16; ++e)
                ap[e] = pw[lrow * NPAD + kk * 32 + a_kidx(e, half)];
            v16h bv0, bv1;
            #pragma unroll
            for (int e = 0; e < 16; ++e) {
                int kd = kk * 32 + e + 16 * half;
                bv0[e] = v_lds[kd * DH + lrow];
                bv1[e] = v_lds[kd * DH + 16 + lrow];
            }
            oc0 = __builtin_amdgcn_wmma_f32_16x16x32_f16(
                false, ap, false, bv0, (short)0, oc0, false, false);
            oc1 = __builtin_amdgcn_wmma_f32_16x16x32_f16(
                false, ap, false, bv1, (short)0, oc1, false, false);
        }

        #pragma unroll
        for (int r = 0; r < 8; ++r) {
            int row = mt * 16 + r + 8 * half;
            if (row < NTOK) {
                size_t g = (size_t)win * NTOK * C_DIM + (size_t)row * C_DIM
                         + (size_t)head * DH;
                ows[g + lrow]      = (_Float16)(oc0[r] * invr[r]);
                ows[g + 16 + lrow] = (_Float16)(oc1[r] * invr[r]);
            }
        }
    }
}

// =====================================================================
// Kernel 4: out = O @ W^T + b, then window-reverse scatter.
// One block per window, 7 waves; WMMA GEMM [224x256] x [256x256]^T.
// Result staged in LDS [C][217] f32, written back with
// GLOBAL_STORE_ASYNC_FROM_LDS_B32 (LDS->memory, no VGPR round trip).
// =====================================================================
__global__ void __launch_bounds__(TPB)
proj_kernel(const _Float16* __restrict__ ows,
            const _Float16* __restrict__ pw16,
            const float* __restrict__ pb,
            float* __restrict__ out) {
    extern __shared__ char smem[];
    float* out_lds = (float*)smem;   // [C_DIM][OUT_LDS_STRIDE]

    int win = blockIdx.x;
    int b = win / NWIN_B;
    int wrem = win - b * NWIN_B;
    int wd = wrem >> 6, wh = (wrem >> 3) & 7, wx = wrem & 7;

    int tid = threadIdx.x;
    int wave = tid >> 5, lane = tid & 31;
    int half = lane >> 4, lrow = lane & 15;

    const _Float16* owin = ows + (size_t)win * NTOK * C_DIM;
    __builtin_prefetch(pw16, 0, 3);   // global_prefetch_b8 (L2-resident weight)

    for (int mt = wave; mt < 14; mt += 7) {
        int tokA = mt * 16 + lrow;
        if (tokA >= NTOK) tokA = 0;       // padded rows: garbage, never stored
        // Preload all 8 A fragments for this M-tile (each O element read once)
        v16h afr[8];
        #pragma unroll
        for (int kk = 0; kk < 8; ++kk)
            #pragma unroll
            for (int e = 0; e < 16; ++e)
                afr[kk][e] = owin[(size_t)tokA * C_DIM + kk * 32 + a_kidx(e, half)];

        for (int nt = 0; nt < 16; ++nt) {
            int ch = nt * 16 + lrow;
            v8f acc = {};
            #pragma unroll
            for (int kk = 0; kk < 8; ++kk) {
                v16h bf;  // B[k][n] = W[n][k]  (x @ W^T)
                #pragma unroll
                for (int e = 0; e < 16; ++e)
                    bf[e] = pw16[(size_t)ch * C_DIM + kk * 32 + 16 * half + e];
                acc = __builtin_amdgcn_wmma_f32_16x16x32_f16(
                    false, afr[kk], false, bf, (short)0, acc, false, false);
            }
            float bias = pb[ch];
            #pragma unroll
            for (int r = 0; r < 8; ++r) {
                int row = mt * 16 + r + 8 * half;
                if (row < NTOK) out_lds[ch * OUT_LDS_STRIDE + row] = acc[r] + bias;
            }
        }
    }
    __syncthreads();

    // window reverse: [C][216] -> [B,C,D,H,W] via async LDS->memory stores.
    // Each thread owns one token (spatial address computed once), loops C.
    if (tid < NTOK) {
        int tok = tid;
        int z = tok / 36; int rem = tok - z * 36;
        int y = rem / 6;  int x = rem - y * 6;
        size_t g0 = (size_t)b * C_DIM * DHW
                  + (size_t)(wd * 6 + z) * (DIM3 * DIM3)
                  + (size_t)(wh * 6 + y) * DIM3 + (wx * 6 + x);
        unsigned l0 = (unsigned)(size_t)out_lds + (unsigned)tok * 4u;
        for (int ch = 0; ch < C_DIM; ++ch) {
            async_store_b32(out + g0 + (size_t)ch * DHW,
                            l0 + (unsigned)(ch * OUT_LDS_STRIDE) * 4u);
        }
    }
    wait_asynccnt0();   // (S_ENDPGM also implies wait-idle)
}

// =====================================================================
// Host launcher
// =====================================================================
extern "C" void kernel_launch(void* const* d_in, const int* in_sizes, int n_in,
                              void* d_out, int out_size, void* d_ws, size_t ws_size,
                              hipStream_t stream) {
    const float* qm  = (const float*)d_in[0];
    const float* km  = (const float*)d_in[1];
    const float* vm  = (const float*)d_in[2];
    const float* wq  = (const float*)d_in[3];
    const float* bq  = (const float*)d_in[4];
    const float* wkv = (const float*)d_in[5];
    const float* bkv = (const float*)d_in[6];
    const float* pwt = (const float*)d_in[7];
    const float* pb  = (const float*)d_in[8];
    float* out = (float*)d_out;

    // workspace layout (f16):  qn | kn | vn | o | proj_w16   (~432 MB)
    const size_t S1 = (size_t)BW * NTOK * C_DIM;  // elements per tensor
    _Float16* qn   = (_Float16*)d_ws;
    _Float16* kn   = qn + S1;
    _Float16* vn   = kn + S1;
    _Float16* ows  = vn + S1;
    _Float16* pw16 = ows + S1;

    // 1) LayerNorm + window partition (q uses ln_q * folded scale)
    dim3 gA((NB * DHW + 255) / 256, 3);
    ln_partition_kernel<<<gA, 256, 0, stream>>>(qm, km, vm, wq, bq, wkv, bkv,
                                                qn, kn, vn);

    // 2) convert projection weight to f16
    cvt_w_kernel<<<(C_DIM * C_DIM + 255) / 256, 256, 0, stream>>>(
        pwt, pw16, C_DIM * C_DIM);

    // 3) attention: 1024 windows x 8 heads, 7 waves/block
    dim3 gB(BW, NH);
    size_t ldsB = (size_t)(3 * NPAD * DH + 7 * 16 * NPAD) * sizeof(_Float16); // ~91 KB
    attn_kernel<<<gB, TPB, ldsB, stream>>>(qn, kn, vn, ows);

    // 4) projection + window reverse, 7 waves/block
    size_t ldsC = (size_t)C_DIM * OUT_LDS_STRIDE * sizeof(float);  // ~222 KB (<320 KB WGP)
    proj_kernel<<<BW, TPB, ldsC, stream>>>(ows, pw16, pb, out);
}
